// Geo_attention_47072841564865
// MI455X (gfx1250) — matrix operand order repaired and checked by reading, not compiled, not run on hardware
//
#include <hip/hip_runtime.h>

// ---------------------------------------------------------------------------
// Geo-attention for MI455X (gfx1250, wave32, WMMA 16x16x32 f16 -> f32 acc,
// async global->LDS staging for conversion-free tile copies)
// N = 20000 points, NN = 32 neighbors, ND = 128, NH = 8 heads
// ---------------------------------------------------------------------------

typedef _Float16 h16v __attribute__((ext_vector_type(16)));
typedef _Float16 h8v  __attribute__((ext_vector_type(8)));
typedef float    f8v  __attribute__((ext_vector_type(8)));

__device__ __forceinline__ f8v wmma16(h16v a, h16v b, f8v c) {
  // D = A(16x32 f16) * B(32x16 f16) + C(16x16 f32)
  return __builtin_amdgcn_wmma_f32_16x16x32_f16(
      /*neg_a=*/false, a, /*neg_b=*/false, b,
      /*c_mod=*/(short)0, c, /*reuse_a=*/false, /*reuse_b=*/false);
}

// Async global->LDS copies (CDNA5 ISA 10.x / 15.18.3; tracked with ASYNCcnt).
// lds_addr is the low 32 bits of the generic pointer (aperture-relative).
__device__ __forceinline__ void async_copy_b128(unsigned lds_addr, const void* g) {
  asm volatile("global_load_async_to_lds_b128 %0, %1, off"
               :: "v"(lds_addr), "v"(g) : "memory");
}
__device__ __forceinline__ void async_copy_b64(unsigned lds_addr, const void* g) {
  asm volatile("global_load_async_to_lds_b64 %0, %1, off"
               :: "v"(lds_addr), "v"(g) : "memory");
}
__device__ __forceinline__ void wait_async0() {
  asm volatile("s_wait_asynccnt 0x0" ::: "memory");
}
__device__ __forceinline__ unsigned lds_off(const void* p) {
  return (unsigned)(unsigned long long)p;
}

// A-matrix fragment, 16x32 f16, row-major source with leading dim `ld`.
// Lanes 0-15 hold row m=lane, K in [0..15];
// lanes 16-31 hold K in [16..31] (matches load_b_frag).
__device__ __forceinline__ h16v load_a_frag(const _Float16* tile, int ld, int lane) {
  const int m  = lane & 15;
  const int kb = (lane >> 4) << 4;        // 0 or 16
  const _Float16* p = tile + m * ld + kb;
  h16v a;
#pragma unroll
  for (int i = 0; i < 8; ++i) a[i] = p[i];
#pragma unroll
  for (int i = 0; i < 8; ++i) a[8 + i] = p[8 + i];
  return a;
}

// B-matrix fragment, 32x16 (KxN) f16. `wt` is the TRANSPOSED weight, i.e.
// row-major (N_out x K_in), so column n of B is the contiguous row n of wt.
// Lanes 0-15: col n=lane, K 0..15; lanes 16-31: col n=lane-16, K 16..31.
__device__ __forceinline__ h16v load_b_frag(const _Float16* wt, int ld, int lane) {
  const int n  = lane & 15;
  const int kb = (lane >> 4) << 4;        // 0 or 16
  const _Float16* p = wt + n * ld + kb;
  h16v b;
#pragma unroll
  for (int i = 0; i < 16; ++i) b[i] = p[i];
  return b;
}

__device__ __forceinline__ float elu_f(float v) {
  return v > 0.0f ? v : (__expf(v) - 1.0f);
}

// ---------------------------------------------------------------------------
// Kernel 0: convert W (K_in x N_out, f32) -> Wt (N_out x K_in, f16)
// ---------------------------------------------------------------------------
__global__ void prep_w_kernel(const float* __restrict__ W, _Float16* __restrict__ Wt,
                              int K, int N) {
  int idx = blockIdx.x * 256 + threadIdx.x;
  if (idx < K * N) {
    int k = idx / N;
    int n = idx - k * N;
    Wt[n * K + k] = (_Float16)W[idx];
  }
}

// ---------------------------------------------------------------------------
// Kernel 1: Q = elu(F @ qW1 + qb1) @ qW2 + qb2     (N x 1024, stored f16)
// 16 rows per block, 8 waves. Layer1: wave w -> col tile w. Layer2: wave w ->
// col tiles w*128 .. w*128+112, with A fragments register-cached (reused 8x).
// ---------------------------------------------------------------------------
__global__ __launch_bounds__(256)
void q_mlp_kernel(const float* __restrict__ feat,
                  const _Float16* __restrict__ wq1t, const float* __restrict__ qb1,
                  const _Float16* __restrict__ wq2t, const float* __restrict__ qb2,
                  _Float16* __restrict__ Qbuf) {
  __shared__ _Float16 Af[16 * 128];
  __shared__ _Float16 Hf[16 * 128];
  const int row0 = blockIdx.x * 16;
  const int tid  = threadIdx.x;
  const int lane = tid & 31;
  const int w    = tid >> 5;

  for (int i = tid; i < 16 * 128; i += 256)
    Af[i] = (_Float16)feat[(size_t)(row0 + (i >> 7)) * 128 + (i & 127)];
  __syncthreads();

  {  // layer 1
    const int n0 = w << 4;
    f8v c = {};
#pragma unroll
    for (int k0 = 0; k0 < 128; k0 += 32) {
      h16v a = load_a_frag(Af + k0, 128, lane);
      h16v b = load_b_frag(wq1t + n0 * 128 + k0, 128, lane);
      c = wmma16(a, b, c);
    }
    const int n  = n0 + (lane & 15);
    const float bs = qb1[n];
#pragma unroll
    for (int r = 0; r < 8; ++r) {
      const int m = r + ((lane >> 4) << 3);
      Hf[m * 128 + n] = (_Float16)elu_f(c[r] + bs);
    }
  }
  __syncthreads();

  {  // layer 2: 1024 cols; A fragments loaded ONCE, reused for 8 col tiles
    h16v afr[4];
#pragma unroll
    for (int k = 0; k < 4; ++k) afr[k] = load_a_frag(Hf + k * 32, 128, lane);
#pragma unroll
    for (int t = 0; t < 8; ++t) {
      const int n0 = (w << 7) + (t << 4);
      f8v c = {};
#pragma unroll
      for (int k = 0; k < 4; ++k) {
        h16v b = load_b_frag(wq2t + n0 * 128 + k * 32, 128, lane);
        c = wmma16(afr[k], b, c);
      }
      const int n  = n0 + (lane & 15);
      const float bs = qb2[n];
#pragma unroll
      for (int r = 0; r < 8; ++r) {
        const int m = r + ((lane >> 4) << 3);
        Qbuf[(size_t)(row0 + m) * 1024 + n] = (_Float16)(c[r] + bs);
      }
    }
  }
}

// (32x128) = f(src(32x128) @ Wt(128x128) + bias); wave w -> col tile w,
// row tiles 0 and 1 sharing register-cached B fragments (B loaded once).
template <bool DO_ELU>
__device__ __forceinline__ void gemm32x128(const _Float16* src, const _Float16* __restrict__ wt,
                                           const float* __restrict__ bias, _Float16* dst,
                                           int lane, int w) {
  const int n0 = w << 4;
  h16v bfr[4];
#pragma unroll
  for (int k = 0; k < 4; ++k) bfr[k] = load_b_frag(wt + n0 * 128 + k * 32, 128, lane);

  const int n  = n0 + (lane & 15);
  const float bs = bias[n];
#pragma unroll
  for (int rt = 0; rt < 2; ++rt) {
    f8v c = {};
#pragma unroll
    for (int k = 0; k < 4; ++k) {
      h16v a = load_a_frag(src + rt * 16 * 128 + k * 32, 128, lane);
      c = wmma16(a, bfr[k], c);
    }
#pragma unroll
    for (int r = 0; r < 8; ++r) {
      const int m = rt * 16 + r + ((lane >> 4) << 3);
      float v = c[r] + bs;
      if (DO_ELU) v = elu_f(v);
      dst[m * 128 + n] = (_Float16)v;
    }
  }
}

// ---------------------------------------------------------------------------
// Kernel 2: one block per point. Geometry -> geo MLP -> K/V MLPs (WMMA) ->
// attention (VALU, ~1.5% of FLOPs) -> attn output (N x 1024 f16).
// Q row staged via async global->LDS copy (conversion-free, 2 KB).
// ---------------------------------------------------------------------------
__global__ __launch_bounds__(256)
void point_kernel(const float* __restrict__ feat, const float* __restrict__ x,
                  const float* __restrict__ nuv, const int* __restrict__ topk,
                  const float* __restrict__ gW1, const float* __restrict__ gb1,
                  const _Float16* __restrict__ wg2t, const float* __restrict__ gb2,
                  const _Float16* __restrict__ wk1t, const float* __restrict__ kb1,
                  const _Float16* __restrict__ wk2t, const float* __restrict__ kb2,
                  const _Float16* __restrict__ wv1t, const float* __restrict__ vb1,
                  const _Float16* __restrict__ wv2t, const float* __restrict__ vb2,
                  const _Float16* __restrict__ Qbuf, _Float16* __restrict__ Abuf) {
  __shared__ _Float16 bufA[32 * 128];  // geo_fea, later V
  __shared__ _Float16 bufB[32 * 128];  // hidden activations
  __shared__ _Float16 bufC[32 * 128];  // K
  __shared__ _Float16 qs[8 * 128];     // Q row (f16, async-staged)
  __shared__ float rl[32][12];
  __shared__ float dis[32];
  __shared__ int   idx[32];
  __shared__ float prob[8][32];
  __shared__ float nuvi[9];
  __shared__ float xi[3];

  const int i    = blockIdx.x;
  const int tid  = threadIdx.x;
  const int lane = tid & 31;
  const int w    = tid >> 5;

  // Async-stage this point's Q row (1024 f16 = 2 KB): 4 halves per thread.
  async_copy_b64(lds_off(qs) + tid * 8, Qbuf + (size_t)i * 1024 + tid * 4);

  if (tid < 9)  nuvi[tid] = nuv[(size_t)i * 9 + tid];
  if (tid < 3)  xi[tid]   = x[(size_t)i * 3 + tid];
  if (tid < 32) idx[tid]  = topk[(size_t)i * 32 + tid];
  wait_async0();
  __syncthreads();

  if (tid < 32) {  // geometry: dis + RL (32 x 12)
    const int t = tid;
    const int j = idx[t];
    const float ox = x[(size_t)j * 3 + 0] - xi[0];
    const float oy = x[(size_t)j * 3 + 1] - xi[1];
    const float oz = x[(size_t)j * 3 + 2] - xi[2];
    dis[t] = __expf(-0.5f * (ox * ox + oy * oy + oz * oz));  // RADIUS = 1
#pragma unroll
    for (int r = 0; r < 3; ++r)
      rl[t][r] = nuvi[r * 3 + 0] * ox + nuvi[r * 3 + 1] * oy + nuvi[r * 3 + 2] * oz;
#pragma unroll
    for (int k = 0; k < 3; ++k) {
      const float a0 = nuv[(size_t)j * 9 + k * 3 + 0];
      const float a1 = nuv[(size_t)j * 9 + k * 3 + 1];
      const float a2 = nuv[(size_t)j * 9 + k * 3 + 2];
#pragma unroll
      for (int r = 0; r < 3; ++r)
        rl[t][3 + k * 3 + r] =
            nuvi[r * 3 + 0] * a0 + nuvi[r * 3 + 1] * a1 + nuvi[r * 3 + 2] * a2;
    }
  }
  __syncthreads();

  {  // geo layer 1 (K=12, VALU): hidden -> bufB
    const int t  = tid >> 3;
    const int d0 = (tid & 7) << 4;
#pragma unroll
    for (int dd = 0; dd < 16; ++dd) {
      const int d = d0 + dd;
      float s = gb1[d];
#pragma unroll
      for (int c = 0; c < 12; ++c) s += rl[t][c] * gW1[c * 128 + d];
      bufB[t * 128 + d] = (_Float16)elu_f(s);
    }
  }
  __syncthreads();

  {  // geo layer 2 + dis * (.) * features_nn  -> bufA (geo_fea)
    const int n0 = w << 4;
    h16v bfr[4];
#pragma unroll
    for (int k = 0; k < 4; ++k) bfr[k] = load_b_frag(wg2t + n0 * 128 + k * 32, 128, lane);
    const int n  = n0 + (lane & 15);
    const float bs = gb2[n];
#pragma unroll
    for (int rt = 0; rt < 2; ++rt) {
      f8v c = {};
#pragma unroll
      for (int k = 0; k < 4; ++k) {
        h16v a = load_a_frag(bufB + rt * 16 * 128 + k * 32, 128, lane);
        c = wmma16(a, bfr[k], c);
      }
#pragma unroll
      for (int r = 0; r < 8; ++r) {
        const int m  = rt * 16 + r + ((lane >> 4) << 3);
        const float fn = feat[(size_t)idx[m] * 128 + n];  // gather, L2-resident
        bufA[m * 128 + n] = (_Float16)(dis[m] * (c[r] + bs) * fn);
      }
    }
  }
  __syncthreads();

  gemm32x128<true >(bufA, wk1t, kb1, bufB, lane, w); __syncthreads();  // K hidden
  gemm32x128<false>(bufB, wk2t, kb2, bufC, lane, w); __syncthreads();  // K
  gemm32x128<true >(bufA, wv1t, vb1, bufB, lane, w); __syncthreads();  // V hidden
  gemm32x128<false>(bufB, wv2t, vb2, bufA, lane, w); __syncthreads();  // V (over geo)

  {  // logits + softmax: wave w = head, lane = neighbor
    const int h = w, t = lane;
    float s = 0.0f;
    const h8v* kp = (const h8v*)(bufC + t * 128);
    const h8v* qp = (const h8v*)(qs + h * 128);
#pragma unroll
    for (int d8 = 0; d8 < 16; ++d8) {
      h8v kv = kp[d8];
      h8v qv = qp[d8];
#pragma unroll
      for (int q = 0; q < 8; ++q) s += (float)qv[q] * (float)kv[q];
    }
    s *= 0.0883883476f;                 // 1/sqrt(128)
    if (idx[t] == 0) s = -1e30f;        // mask
    float mx = s;
#pragma unroll
    for (int o = 16; o; o >>= 1) mx = fmaxf(mx, __shfl_xor(mx, o, 32));
    const float e = __expf(s - mx);
    float sm = e;
#pragma unroll
    for (int o = 16; o; o >>= 1) sm += __shfl_xor(sm, o, 32);
    prob[h][t] = e / sm;
  }
  __syncthreads();

  {  // out = prob @ V -> Abuf (f16)
    const int h  = tid >> 5;
    const int d0 = (tid & 31) << 2;
    float acc[4] = {0.f, 0.f, 0.f, 0.f};
    for (int t = 0; t < 32; ++t) {
      const float p = prob[h][t];
#pragma unroll
      for (int q = 0; q < 4; ++q) acc[q] += p * (float)bufA[t * 128 + d0 + q];
    }
#pragma unroll
    for (int q = 0; q < 4; ++q)
      Abuf[(size_t)i * 1024 + h * 128 + d0 + q] = (_Float16)acc[q];
  }
}

// ---------------------------------------------------------------------------
// Kernel 3: decode MLP (1024->128->128) + residual + LayerNorm. 16 rows/block.
// The 16x1024 attention tile (32 KB, contiguous) is staged once into LDS via
// async global->LDS b128 copies, removing the 8x redundant per-wave global
// fragment loads.
// ---------------------------------------------------------------------------
__global__ __launch_bounds__(256)
void decode_kernel(const _Float16* __restrict__ Abuf,
                   const _Float16* __restrict__ wd1t, const float* __restrict__ db1,
                   const _Float16* __restrict__ wd2t, const float* __restrict__ db2,
                   const float* __restrict__ feat,
                   const float* __restrict__ lng, const float* __restrict__ lnb,
                   float* __restrict__ out) {
  __shared__ _Float16 Asub[16 * 1024];  // 32 KB, async-staged
  __shared__ _Float16 Hf[16 * 128];
  __shared__ float    Of[16 * 128];
  const int row0 = blockIdx.x * 16;
  const int tid  = threadIdx.x;
  const int lane = tid & 31;
  const int w    = tid >> 5;

  {  // stage A tile: 16384 halves; 8 b128 copies (8 halves each) per thread
    const _Float16* gsrc = Abuf + (size_t)row0 * 1024;
    const unsigned  lb   = lds_off(Asub);
#pragma unroll
    for (int c = 0; c < 8; ++c) {
      const int e = tid * 8 + c * 2048;           // element (half) offset
      async_copy_b128(lb + e * 2, gsrc + e);
    }
    wait_async0();
  }
  __syncthreads();

  {  // layer 1: K = 1024, A fragments from LDS
    const int n0 = w << 4;
    f8v c = {};
    for (int k0 = 0; k0 < 1024; k0 += 32) {
      h16v a = load_a_frag(Asub + k0, 1024, lane);
      h16v b = load_b_frag(wd1t + n0 * 1024 + k0, 1024, lane);
      c = wmma16(a, b, c);
    }
    const int n  = n0 + (lane & 15);
    const float bs = db1[n];
#pragma unroll
    for (int r = 0; r < 8; ++r) {
      const int m = r + ((lane >> 4) << 3);
      Hf[m * 128 + n] = (_Float16)elu_f(c[r] + bs);
    }
  }
  __syncthreads();

  {  // layer 2 + residual
    const int n0 = w << 4;
    f8v c = {};
#pragma unroll
    for (int k0 = 0; k0 < 128; k0 += 32) {
      h16v a = load_a_frag(Hf + k0, 128, lane);
      h16v b = load_b_frag(wd2t + n0 * 128 + k0, 128, lane);
      c = wmma16(a, b, c);
    }
    const int n  = n0 + (lane & 15);
    const float bs = db2[n];
#pragma unroll
    for (int r = 0; r < 8; ++r) {
      const int m = r + ((lane >> 4) << 3);
      Of[m * 128 + n] = c[r] + bs + feat[(size_t)(row0 + m) * 128 + n];
    }
  }
  __syncthreads();

  // LayerNorm: wave w handles rows 2w and 2w+1
#pragma unroll
  for (int rr = 0; rr < 2; ++rr) {
    const int m = (w << 1) + rr;
    float s = 0.f, s2 = 0.f;
#pragma unroll
    for (int d = lane; d < 128; d += 32) {
      const float v = Of[m * 128 + d];
      s += v; s2 += v * v;
    }
#pragma unroll
    for (int o = 16; o; o >>= 1) { s += __shfl_xor(s, o, 32); s2 += __shfl_xor(s2, o, 32); }
    const float mu  = s * (1.0f / 128.0f);
    const float var = s2 * (1.0f / 128.0f) - mu * mu;
    const float inv = rsqrtf(var + 1e-5f);
#pragma unroll
    for (int d = lane; d < 128; d += 32)
      out[(size_t)(row0 + m) * 128 + d] = (Of[m * 128 + d] - mu) * inv * lng[d] + lnb[d];
  }
}

// ---------------------------------------------------------------------------
extern "C" void kernel_launch(void* const* d_in, const int* in_sizes, int n_in,
                              void* d_out, int out_size, void* d_ws, size_t ws_size,
                              hipStream_t stream) {
  const float* feat = (const float*)d_in[0];
  const float* x    = (const float*)d_in[1];
  const float* nuv  = (const float*)d_in[2];
  const int*   topk = (const int*)d_in[3];
  const float* qW1 = (const float*)d_in[4];  const float* qb1 = (const float*)d_in[5];
  const float* qW2 = (const float*)d_in[6];  const float* qb2 = (const float*)d_in[7];
  const float* gW1 = (const float*)d_in[8];  const float* gb1 = (const float*)d_in[9];
  const float* gW2 = (const float*)d_in[10]; const float* gb2 = (const float*)d_in[11];
  const float* kW1 = (const float*)d_in[12]; const float* kb1 = (const float*)d_in[13];
  const float* kW2 = (const float*)d_in[14]; const float* kb2 = (const float*)d_in[15];
  const float* vW1 = (const float*)d_in[16]; const float* vb1 = (const float*)d_in[17];
  const float* vW2 = (const float*)d_in[18]; const float* vb2 = (const float*)d_in[19];
  const float* dW1 = (const float*)d_in[20]; const float* db1 = (const float*)d_in[21];
  const float* dW2 = (const float*)d_in[22]; const float* db2 = (const float*)d_in[23];
  const float* lng = (const float*)d_in[24]; const float* lnb = (const float*)d_in[25];
  const int N = in_sizes[0] / 128;

  _Float16* p = (_Float16*)d_ws;
  _Float16* wq1t = p; p += 128 * 128;
  _Float16* wq2t = p; p += 1024 * 128;
  _Float16* wg2t = p; p += 128 * 128;
  _Float16* wk1t = p; p += 128 * 128;
  _Float16* wk2t = p; p += 128 * 128;
  _Float16* wv1t = p; p += 128 * 128;
  _Float16* wv2t = p; p += 128 * 128;
  _Float16* wd1t = p; p += 128 * 1024;
  _Float16* wd2t = p; p += 128 * 128;
  _Float16* Qbuf = p; p += (size_t)N * 1024;
  _Float16* Abuf = p; p += (size_t)N * 1024;

  struct WDesc { const float* W; _Float16* Wt; int K, N; };
  const WDesc wd[9] = {
      {qW1, wq1t, 128, 128},  {qW2, wq2t, 128, 1024}, {gW2, wg2t, 128, 128},
      {kW1, wk1t, 128, 128},  {kW2, wk2t, 128, 128},  {vW1, wv1t, 128, 128},
      {vW2, wv2t, 128, 128},  {dW1, wd1t, 1024, 128}, {dW2, wd2t, 128, 128}};
  for (int t = 0; t < 9; ++t)
    prep_w_kernel<<<(wd[t].K * wd[t].N + 255) / 256, 256, 0, stream>>>(
        wd[t].W, wd[t].Wt, wd[t].K, wd[t].N);

  q_mlp_kernel<<<N / 16, 256, 0, stream>>>(feat, wq1t, qb1, wq2t, qb2, Qbuf);
  point_kernel<<<N, 256, 0, stream>>>(feat, x, nuv, topk, gW1, gb1, wg2t, gb2,
                                      wk1t, kb1, wk2t, kb2, wv1t, vb1, wv2t, vb2,
                                      Qbuf, Abuf);
  decode_kernel<<<N / 16, 256, 0, stream>>>(Abuf, wd1t, db1, wd2t, db2, feat,
                                            lng, lnb, (float*)d_out);
}
